// MLAttention_40407052320889
// MI455X (gfx1250) — matrix operand order repaired and checked by reading, not compiled, not run on hardware
//
#include <hip/hip_runtime.h>
#include <cstdint>
#include <cstddef>

typedef __attribute__((ext_vector_type(16))) __bf16 v16bf;
typedef __attribute__((ext_vector_type(8)))  __bf16 v8bf;
typedef __attribute__((ext_vector_type(8)))  float  v8f;

namespace {
constexpr int E_DIM   = 2048;
constexpr int H_DIM   = 16;
constexpr int D_NOPE  = 128;
constexpr int D_ROPE  = 64;
constexpr int D_QK    = 192;                    // 128 + 64
constexpr int D_V     = 128;
constexpr int Q_LORA  = 1536;
constexpr int KV_LORA = 512;
constexpr int KVA_N   = KV_LORA + D_ROPE;       // 576
constexpr int B_DIM   = 2;
constexpr int S_DIM   = 2048;
constexpr int M_ROWS  = B_DIM * S_DIM;          // 4096
constexpr int QK_N    = H_DIM * D_QK;           // 3072
constexpr int KVB_N   = H_DIM * (D_NOPE + D_V); // 4096
constexpr int ATTN_N  = H_DIM * D_V;            // 2048
}

__device__ inline v8f wmma_bf16(v16bf a, v16bf b, v8f c) {
    return __builtin_amdgcn_wmma_f32_16x16x32_bf16(false, a, false, b, (short)0, c,
                                                   false, false);
}

// --- CDNA5 async global->LDS copy (VGLOBAL encoding: VDST = LDS byte offset,
// --- VADDR = 64-bit global address, no SADDR). Tracked by ASYNCcnt.
__device__ inline void async_copy_b128(unsigned lds_off, const __bf16* gaddr) {
    asm volatile("global_load_async_to_lds_b128 %0, %1, off"
                 :: "v"(lds_off), "v"((unsigned long long)(size_t)gaddr)
                 : "memory");
}
__device__ inline void wait_async0() {
#if __has_builtin(__builtin_amdgcn_s_wait_asynccnt)
    __builtin_amdgcn_s_wait_asynccnt(0);
#else
    asm volatile("s_wait_asynccnt 0" ::: "memory");
#endif
}

// ---------------------------------------------------------------------------
// fp32 -> bf16 streaming convert
// ---------------------------------------------------------------------------
__global__ __launch_bounds__(256)
void f32_to_bf16_kernel(const float* __restrict__ in, __bf16* __restrict__ out, size_t n)
{
    size_t stride = (size_t)gridDim.x * blockDim.x;
    for (size_t i = (size_t)blockIdx.x * blockDim.x + threadIdx.x; i < n; i += stride)
        out[i] = (__bf16)in[i];
}

// ---------------------------------------------------------------------------
// fp32 [R,C] -> bf16 transposed [C,R] via LDS tile (weights become Bt = N x K)
// ---------------------------------------------------------------------------
__global__ __launch_bounds__(256)
void transpose_f32_to_bf16(const float* __restrict__ in, __bf16* __restrict__ out,
                           int R, int C)
{
    __shared__ __bf16 tile[32][33];
    int tx = threadIdx.x & 31, ty = threadIdx.x >> 5;   // 32 x 8 threads
    int c0 = blockIdx.x * 32, r0 = blockIdx.y * 32;
    for (int i = 0; i < 32; i += 8) {
        int r = r0 + ty + i, c = c0 + tx;
        if (r < R && c < C) tile[ty + i][tx] = (__bf16)in[(size_t)r * C + c];
    }
    __syncthreads();
    for (int i = 0; i < 32; i += 8) {
        int c = c0 + ty + i, r = r0 + tx;
        if (r < R && c < C) out[(size_t)c * R + r] = tile[tx][ty + i];
    }
}

// ---------------------------------------------------------------------------
// WMMA GEMM: C[M,N] = A[M,K](bf16,row) * Bt[N,K](bf16,row)^T
// block = 128 threads = 4 waves; wave -> 32x64 tile (2 A-frags share 4 B-frags
// -> 1.5 b128 loads per WMMA); grid = (N/64, M/128)
// A-frag: lane row = m, elem j -> k = 16*half + j
// B-frag: lane col = n, elem j -> k = 16*half + j  (contiguous in Bt)
// ---------------------------------------------------------------------------
template <bool F32OUT>
__global__ __launch_bounds__(128)
void gemm_wmma_kernel(const __bf16* __restrict__ A, const __bf16* __restrict__ Bt,
                      void* __restrict__ Cptr, int K, int lda, int ldb, int ldc)
{
    const int lane = threadIdx.x & 31;
    const int wave = threadIdx.x >> 5;
    const int half = lane >> 4;
    const int l16  = lane & 15;
    const int m_base = blockIdx.y * 128 + wave * 32;
    const int n_base = blockIdx.x * 64;

    v8f zero = {};
    v8f acc[2][4];
    for (int u = 0; u < 2; ++u)
        for (int t = 0; t < 4; ++t) acc[u][t] = zero;

    const __bf16* arow0 = A + (size_t)(m_base + l16) * lda;
    const __bf16* arow1 = A + (size_t)(m_base + 16 + l16) * lda;
    for (int kk = 0; kk < K; kk += 32) {
        v16bf a0, a1;
        {
            v8bf lo0 = *(const v8bf*)(arow0 + kk + 16 * half);
            v8bf hi0 = *(const v8bf*)(arow0 + kk + 16 * half + 8);
            v8bf lo1 = *(const v8bf*)(arow1 + kk + 16 * half);
            v8bf hi1 = *(const v8bf*)(arow1 + kk + 16 * half + 8);
            for (int i = 0; i < 8; ++i) {
                a0[i] = lo0[i]; a0[i + 8] = hi0[i];
                a1[i] = lo1[i]; a1[i + 8] = hi1[i];
            }
        }
        for (int t = 0; t < 4; ++t) {
            const __bf16* brow =
                Bt + (size_t)(n_base + 16 * t + l16) * ldb + kk + 16 * half;
            v16bf b = *(const v16bf*)brow;   // two global_load_b128
            acc[0][t] = wmma_bf16(a0, b, acc[0][t]);
            acc[1][t] = wmma_bf16(a1, b, acc[1][t]);
        }
    }

    float*  Cf = (float*)Cptr;
    __bf16* Cb = (__bf16*)Cptr;
    for (int u = 0; u < 2; ++u)
        for (int t = 0; t < 4; ++t)
            for (int r = 0; r < 8; ++r) {
                size_t idx = (size_t)(m_base + u * 16 + r + 8 * half) * ldc
                           + n_base + 16 * t + l16;
                if (F32OUT) Cf[idx] = acc[u][t][r];
                else        Cb[idx] = (__bf16)acc[u][t][r];
            }
}

// ---------------------------------------------------------------------------
// RMSNorm over `len` leading columns of each row (bf16 in, bf16 out)
// ---------------------------------------------------------------------------
__global__ __launch_bounds__(256)
void rmsnorm_kernel(const __bf16* __restrict__ in, const float* __restrict__ scale,
                    __bf16* __restrict__ out, int in_stride, int out_stride, int len)
{
    __shared__ float red[256];
    const __bf16* x = in + (size_t)blockIdx.x * in_stride;
    float s = 0.f;
    for (int c = threadIdx.x; c < len; c += 256) {
        float v = (float)x[c];
        s += v * v;
    }
    red[threadIdx.x] = s;
    __syncthreads();
    for (int o = 128; o > 0; o >>= 1) {
        if ((int)threadIdx.x < o) red[threadIdx.x] += red[threadIdx.x + o];
        __syncthreads();
    }
    float inv = rsqrtf(red[0] / (float)len + 1e-6f);
    __bf16* y = out + (size_t)blockIdx.x * out_stride;
    for (int c = threadIdx.x; c < len; c += 256)
        y[c] = (__bf16)((float)x[c] * inv * scale[c]);
}

// ---------------------------------------------------------------------------
// Assemble bf16 q [bs,h,192] (rope on last 64), k [bs,h,192] (k_nope | roped
// shared k_rope), and V transposed v_t [b,h,d_v,s].  grid=(B*S,H), block=192
// ---------------------------------------------------------------------------
__global__ __launch_bounds__(192)
void assemble_kernel(const __bf16* __restrict__ q_raw, const __bf16* __restrict__ kvexp,
                     const __bf16* __restrict__ kvc, const int* __restrict__ positions,
                     __bf16* __restrict__ q_bf, __bf16* __restrict__ k_bf,
                     __bf16* __restrict__ v_t)
{
    const int bs = blockIdx.x;
    const int h  = blockIdx.y;
    const int d  = threadIdx.x;          // 0..191
    const int b  = bs / S_DIM;
    const int s  = bs - b * S_DIM;
    const float pos = (float)positions[bs];
    const float NEG_LN_THETA_OVER_32 = -0.28782313662425572f;   // -ln(1e4)/32

    const size_t qbase = (size_t)bs * QK_N + (size_t)h * D_QK;

    float qv, kv;
    if (d < D_NOPE) {
        qv = (float)q_raw[qbase + d];
        kv = (float)kvexp[(size_t)bs * KVB_N + (size_t)h * (D_NOPE + D_V) + d];
    } else {
        int  i  = (d - D_NOPE) & 31;
        bool hi = (d - D_NOPE) >= 32;
        float ang = pos * __expf(NEG_LN_THETA_OVER_32 * (float)i);
        float sn, cs;
        __sincosf(ang, &sn, &cs);
        float q1 = (float)q_raw[qbase + D_NOPE + i];
        float q2 = (float)q_raw[qbase + D_NOPE + 32 + i];
        qv = hi ? (q2 * cs + q1 * sn) : (q1 * cs - q2 * sn);
        float k1 = (float)kvc[(size_t)bs * KVA_N + KV_LORA + i];
        float k2 = (float)kvc[(size_t)bs * KVA_N + KV_LORA + 32 + i];
        kv = hi ? (k2 * cs + k1 * sn) : (k1 * cs - k2 * sn);
    }
    q_bf[qbase + d] = (__bf16)qv;
    k_bf[qbase + d] = (__bf16)kv;

    if (d < D_V)
        v_t[((size_t)(b * H_DIM + h) * D_V + d) * S_DIM + s] =
            kvexp[(size_t)bs * KVB_N + (size_t)h * (D_NOPE + D_V) + D_NOPE + d];
}

// ---------------------------------------------------------------------------
// Flash attention: 4 waves/block, each wave a 16-row q tile (64 q rows/block).
// K/V tiles for each 32-wide causal chunk are double-buffered in LDS and
// filled with global_load_async_to_lds_b128; ASYNCcnt-pipelined prefetch of
// chunk ck+1 overlaps WMMA compute of chunk ck.
// ---------------------------------------------------------------------------
__global__ __launch_bounds__(128)
void attn_kernel(const __bf16* __restrict__ q_bf, const __bf16* __restrict__ k_bf,
                 const __bf16* __restrict__ v_t, __bf16* __restrict__ attn)
{
    const int tid  = threadIdx.x;
    const int lane = tid & 31;
    const int wave = tid >> 5;
    const int half = lane >> 4;
    const int l16  = lane & 15;
    const int h    = blockIdx.y;
    const int b    = blockIdx.z;
    const int q0_blk = blockIdx.x * 64;
    const int q0   = q0_blk + wave * 16;

    __shared__ alignas(32) __bf16 Kt[2][32 * D_QK];   // [krel][d]      12KB x2
    __shared__ alignas(32) __bf16 Vt[2][D_V * 32];    // [dv][krel]      8KB x2
    __shared__ alignas(32) __bf16 Plds[4][16 * 32];   // per-wave P tile

    const int nch_w   = (q0 + 16 + 31) >> 5;          // this wave's chunk count
    const int nch_max = (q0_blk + 64 + 31) >> 5;      // block chunk count

    // Q tile -> 6 A-fragments (d = 0..191), loaded once from global
    v16bf qf[6];
    {
        const __bf16* qrow =
            q_bf + ((size_t)(b * S_DIM + q0 + l16) * H_DIM + h) * D_QK;
        for (int c = 0; c < 6; ++c) {
            v8bf lo = *(const v8bf*)(qrow + c * 32 + 16 * half);
            v8bf hi = *(const v8bf*)(qrow + c * 32 + 16 * half + 8);
            for (int i = 0; i < 8; ++i) { qf[c][i] = lo[i]; qf[c][i + 8] = hi[i]; }
        }
    }

    v8f zero = {};
    v8f O[8];
    for (int t = 0; t < 8; ++t) O[t] = zero;
    float mrow[8], lrow[8];
    for (int r = 0; r < 8; ++r) { mrow[r] = -1e30f; lrow[r] = 0.f; }

    const size_t krow_stride = (size_t)H_DIM * D_QK;       // 3072
    const __bf16* kbase = k_bf + (size_t)b * S_DIM * krow_stride + (size_t)h * D_QK;
    const __bf16* vbase = v_t + (size_t)(b * H_DIM + h) * D_V * S_DIM;

    // async prefetch of one 32-wide K/V chunk into LDS buffer `buf`
    auto prefetch = [&](int kb, int buf) {
        const unsigned koff = (unsigned)(size_t)&Kt[buf][0];
        const unsigned voff = (unsigned)(size_t)&Vt[buf][0];
        // K tile: 32 x 192 bf16 = 12288B -> 768 16B ops (24 per row), 6/thread
        for (int r = 0; r < 6; ++r) {
            int op  = tid + 128 * r;
            int row = op / 24;
            int col = (op % 24) * 8;
            async_copy_b128(koff + (unsigned)((row * D_QK + col) * 2),
                            kbase + (size_t)(kb + row) * krow_stride + col);
        }
        // V tile: 128 x 32 bf16 = 8192B -> 512 16B ops (4 per row), 4/thread
        for (int r = 0; r < 4; ++r) {
            int op  = tid + 128 * r;
            int row = op / 4;
            int col = (op % 4) * 8;
            async_copy_b128(voff + (unsigned)((row * 32 + col) * 2),
                            vbase + (size_t)row * S_DIM + kb + col);
        }
    };

    const float sc = 0.07216878364870323f;   // 1/sqrt(192)

    prefetch(0, 0);
    for (int ck = 0; ck < nch_max; ++ck) {
        wait_async0();           // chunk ck's async copies complete (in-order)
        __syncthreads();         // all waves done with buf[(ck+1)&1] reads
        if (ck + 1 < nch_max) prefetch((ck + 1) * 32, (ck + 1) & 1);

        if (ck < nch_w) {
            const int kb = ck * 32;
            const int pb = ck & 1;
            v8f s0 = zero, s1 = zero;
            {
                const __bf16* k0 = &Kt[pb][0] + l16 * D_QK + 16 * half;
                const __bf16* k1 = &Kt[pb][0] + (16 + l16) * D_QK + 16 * half;
                for (int c = 0; c < 6; ++c) {
                    v16bf kf0 = *(const v16bf*)(k0 + c * 32);   // ds_load_b128 x2
                    v16bf kf1 = *(const v16bf*)(k1 + c * 32);
                    s0 = wmma_bf16(qf[c], kf0, s0);
                    s1 = wmma_bf16(qf[c], kf1, s1);
                }
            }
            const bool last = (ck == nch_w - 1);
            float p0[8], p1[8], corr[8];
            for (int r = 0; r < 8; ++r) {
                const int m = r + 8 * half;
                float a  = s0[r] * sc;
                float bv = s1[r] * sc;
                if (last) {                 // causal mask lives in last chunk only
                    if (kb + l16 > q0 + m)      a  = -1e30f;
                    if (kb + 16 + l16 > q0 + m) bv = -1e30f;
                }
                float mx = fmaxf(a, bv);
                for (int o = 1; o < 16; o <<= 1) mx = fmaxf(mx, __shfl_xor(mx, o, 32));
                float mnew = fmaxf(mrow[r], mx);
                corr[r] = __expf(mrow[r] - mnew);
                mrow[r] = mnew;
                float e0 = __expf(a - mnew), e1 = __expf(bv - mnew);
                float rs = e0 + e1;
                for (int o = 1; o < 16; o <<= 1) rs += __shfl_xor(rs, o, 32);
                lrow[r] = lrow[r] * corr[r] + rs;
                p0[r] = e0; p1[r] = e1;
            }
            for (int t = 0; t < 8; ++t)
                for (int r = 0; r < 8; ++r) O[t][r] *= corr[r];

            // C-layout -> LDS -> A-layout for P (same-wave LDS ops are in-order)
            for (int r = 0; r < 8; ++r) {
                const int m = r + 8 * half;
                Plds[wave][m * 32 + l16]      = (__bf16)p0[r];
                Plds[wave][m * 32 + 16 + l16] = (__bf16)p1[r];
            }
            v16bf pf;
            {
                const v8bf* prow = (const v8bf*)(&Plds[wave][0] + l16 * 32);
                v8bf plo = prow[2 * half];
                v8bf phi = prow[2 * half + 1];
                for (int i = 0; i < 8; ++i) { pf[i] = plo[i]; pf[i + 8] = phi[i]; }
            }
            for (int t = 0; t < 8; ++t) {
                const __bf16* vrow = &Vt[pb][0] + (t * 16 + l16) * 32 + 16 * half;
                v16bf vf = *(const v16bf*)vrow;
                O[t] = wmma_bf16(pf, vf, O[t]);
            }
        }
    }

    for (int t = 0; t < 8; ++t)
        for (int r = 0; r < 8; ++r) {
            const int m = r + 8 * half;
            float val = O[t][r] / lrow[r];
            attn[((size_t)(b * S_DIM + q0 + m) * H_DIM + h) * D_V + t * 16 + l16] =
                (__bf16)val;
        }
}

// ---------------------------------------------------------------------------
// host launcher
// ---------------------------------------------------------------------------
extern "C" void kernel_launch(void* const* d_in, const int* in_sizes, int n_in,
                              void* d_out, int out_size, void* d_ws, size_t ws_size,
                              hipStream_t stream)
{
    (void)in_sizes; (void)n_in; (void)out_size; (void)ws_size;
    const float* x         = (const float*)d_in[0];
    const int*   positions = (const int*)d_in[1];
    /* d_in[2] = mask: causal mask applied analytically in attn_kernel */
    const float* Wqa       = (const float*)d_in[3];
    const float* qa_scale  = (const float*)d_in[4];
    const float* Wqb       = (const float*)d_in[5];
    const float* Wkva      = (const float*)d_in[6];
    const float* kva_scale = (const float*)d_in[7];
    const float* Wkvb      = (const float*)d_in[8];
    const float* Wout      = (const float*)d_in[9];
    float* out = (float*)d_out;

    char* base = (char*)d_ws;
    size_t off = 0;
    auto alloc = [&](size_t elems) -> __bf16* {
        __bf16* p = (__bf16*)(base + off);
        off = (off + elems * sizeof(__bf16) + 255) & ~(size_t)255;
        return p;
    };

    __bf16* x_bf    = alloc((size_t)M_ROWS * E_DIM);
    __bf16* WqaT    = alloc((size_t)Q_LORA * E_DIM);
    __bf16* WkvaT   = alloc((size_t)KVA_N * E_DIM);
    __bf16* WqbT    = alloc((size_t)QK_N * Q_LORA);
    __bf16* WkvbT   = alloc((size_t)KVB_N * KV_LORA);
    __bf16* WoutT   = alloc((size_t)E_DIM * ATTN_N);
    __bf16* qa_raw  = alloc((size_t)M_ROWS * Q_LORA);
    __bf16* kvc_raw = alloc((size_t)M_ROWS * KVA_N);
    __bf16* qa_n    = alloc((size_t)M_ROWS * Q_LORA);
    __bf16* kvl_n   = alloc((size_t)M_ROWS * KV_LORA);
    __bf16* q_raw   = alloc((size_t)M_ROWS * QK_N);
    __bf16* kvexp   = alloc((size_t)M_ROWS * KVB_N);
    __bf16* q_bf    = alloc((size_t)M_ROWS * QK_N);
    __bf16* k_bf    = alloc((size_t)M_ROWS * QK_N);
    __bf16* v_t     = alloc((size_t)B_DIM * H_DIM * D_V * S_DIM);
    __bf16* attn    = alloc((size_t)M_ROWS * ATTN_N);

    // --- precision conversion + weight transposes (Bt = N x K, bf16) ---
    f32_to_bf16_kernel<<<4096, 256, 0, stream>>>(x, x_bf, (size_t)M_ROWS * E_DIM);
    transpose_f32_to_bf16<<<dim3(Q_LORA / 32, E_DIM / 32), 256, 0, stream>>>(Wqa,  WqaT,  E_DIM,  Q_LORA);
    transpose_f32_to_bf16<<<dim3(KVA_N / 32,  E_DIM / 32), 256, 0, stream>>>(Wkva, WkvaT, E_DIM,  KVA_N);
    transpose_f32_to_bf16<<<dim3(QK_N / 32,  Q_LORA / 32), 256, 0, stream>>>(Wqb,  WqbT,  Q_LORA, QK_N);
    transpose_f32_to_bf16<<<dim3(KVB_N / 32, KV_LORA / 32), 256, 0, stream>>>(Wkvb, WkvbT, KV_LORA, KVB_N);
    transpose_f32_to_bf16<<<dim3(E_DIM / 32,  ATTN_N / 32), 256, 0, stream>>>(Wout, WoutT, ATTN_N, E_DIM);

    // --- down projections ---
    gemm_wmma_kernel<false><<<dim3(Q_LORA / 64, M_ROWS / 128), 128, 0, stream>>>(
        x_bf, WqaT, qa_raw, E_DIM, E_DIM, E_DIM, Q_LORA);
    gemm_wmma_kernel<false><<<dim3(KVA_N / 64, M_ROWS / 128), 128, 0, stream>>>(
        x_bf, WkvaT, kvc_raw, E_DIM, E_DIM, E_DIM, KVA_N);

    // --- RMSNorms ---
    rmsnorm_kernel<<<M_ROWS, 256, 0, stream>>>(qa_raw, qa_scale, qa_n,
                                               Q_LORA, Q_LORA, Q_LORA);
    rmsnorm_kernel<<<M_ROWS, 256, 0, stream>>>(kvc_raw, kva_scale, kvl_n,
                                               KVA_N, KV_LORA, KV_LORA);

    // --- up projections ---
    gemm_wmma_kernel<false><<<dim3(QK_N / 64, M_ROWS / 128), 128, 0, stream>>>(
        qa_n, WqbT, q_raw, Q_LORA, Q_LORA, Q_LORA, QK_N);
    gemm_wmma_kernel<false><<<dim3(KVB_N / 64, M_ROWS / 128), 128, 0, stream>>>(
        kvl_n, WkvbT, kvexp, KV_LORA, KV_LORA, KV_LORA, KVB_N);

    // --- RoPE + q/k/v assembly ---
    assemble_kernel<<<dim3(M_ROWS, H_DIM), 192, 0, stream>>>(
        q_raw, kvexp, kvc_raw, positions, q_bf, k_bf, v_t);

    // --- flash attention (async-LDS double-buffered K/V) ---
    attn_kernel<<<dim3(S_DIM / 64, H_DIM, B_DIM), 128, 0, stream>>>(
        q_bf, k_bf, v_t, attn);

    // --- output projection (fp32 result) ---
    gemm_wmma_kernel<true><<<dim3(E_DIM / 64, M_ROWS / 128), 128, 0, stream>>>(
        attn, WoutT, out, ATTN_N, ATTN_N, ATTN_N, E_DIM);
}
